// LePEAttention_2121713844305
// MI455X (gfx1250) — matrix operand order repaired and checked by reading, not compiled, not run on hardware
//
#include <hip/hip_runtime.h>
#include <stdint.h>

// ---------------- problem constants ----------------
#define RES       64
#define SPLIT     8
#define DIM       512
#define NUM_HEADS 8
#define HEAD_DIM  64
#define WS        512            // tokens per window (64 x 8)
#define NWIN      8              // windows per batch image
#define BATCH     8
#define L_TOK     (RES * RES)
#define QK_SCALE  0.125f         // HEAD_DIM^-0.5

// ---------------- LDS layout (dwords) ----------------
#define KSTRIDE   36u                       // 64 bf16 (32 dw) + 4 dw pad; 144B row, 16B aligned
#define VTSTRIDE  260u                      // 512 bf16 (256 dw) + 4 dw pad; 1040B row
#define K_DW      (WS * KSTRIDE)            // 18432 dw
#define VT_DW     (HEAD_DIM * VTSTRIDE)     // 16640 dw
#define P_DW      (8 * 256)                 // per-wave 16x32 bf16 scratch (256 dw each)
#define W_F       (HEAD_DIM * 10)           // 9 weights + bias per channel
#define SMEM_BYTES ((K_DW + VT_DW + P_DW + W_F) * 4)   // 151,040 B

typedef __attribute__((ext_vector_type(16))) __bf16   v16bf;
typedef __attribute__((ext_vector_type(8)))  float    v8f;
typedef __attribute__((ext_vector_type(4)))  uint32_t u32x4;   // trivial 16B vector (union-safe)

union Frag {                      // one WMMA 16-bit A/B fragment: 8 VGPRs = 16 bf16
    uint32_t u[8];
    u32x4    x[2];
    v16bf    v;
};

// single-instruction packed f32 -> 2 x bf16 (RNE) conversion
__device__ __forceinline__ uint32_t pack_bf16_2(float a, float b) {
    uint32_t r;
    asm("v_cvt_pk_bf16_f32 %0, %1, %2" : "=v"(r) : "v"(a), "v"(b));
    return r;
}

// one-instruction max (avoids LLVM's canonicalization self-max on intrinsic outputs)
__device__ __forceinline__ float vmaxf(float a, float b) {
    float r;
    asm("v_max_num_f32 %0, %1, %2" : "=v"(r) : "v"(a), "v"(b));
    return r;
}

// one-instruction xor-butterfly step: dst = max(dpp_xmask(src), src).
__device__ __forceinline__ float rowmax16(float v) {
    float t;
    asm("v_max_num_f32_dpp %0, %1, %1 row_xmask:8 row_mask:0xf bank_mask:0xf bound_ctrl:1"
        : "=&v"(t) : "v"(v));
    asm("v_max_num_f32_dpp %0, %1, %1 row_xmask:4 row_mask:0xf bank_mask:0xf bound_ctrl:1"
        : "=&v"(v) : "v"(t));
    asm("v_max_num_f32_dpp %0, %1, %1 row_xmask:2 row_mask:0xf bank_mask:0xf bound_ctrl:1"
        : "=&v"(t) : "v"(v));
    asm("v_max_num_f32_dpp %0, %1, %1 row_xmask:1 row_mask:0xf bank_mask:0xf bound_ctrl:1"
        : "=&v"(v) : "v"(t));
    return v;
}

__global__ __launch_bounds__(256)
void cswin_lepe_attn(const float* __restrict__ qkv,
                     const float* __restrict__ conv_w,
                     const float* __restrict__ conv_b,
                     float* __restrict__ out)
{
    extern __shared__ uint32_t smem[];
    uint32_t* sK  = smem;                 // bf16 K,  [512 tok][64 d], row stride KSTRIDE dw
    uint32_t* sVT = smem + K_DW;          // bf16 V^T,[64 d][512 tok], row stride VTSTRIDE dw
    uint32_t* sP  = sVT + VT_DW;          // per-wave P scratch, 8 x [16][32] bf16
    float*    sW  = (float*)(sP + P_DW);  // [64][10] conv weights+bias for this head

    const int wg   = blockIdx.x;          // 0..511
    const int head = wg & 7;
    const int win  = (wg >> 3) & 7;
    const int b    = wg >> 6;

    const int tid    = threadIdx.x;
    const int lane   = tid & 31;
    const int wave   = tid >> 5;
    const int laneLo = lane & 15;         // N column / A row within 16
    const int laneHi = lane >> 4;         // which 16-lane half

    const float* qb  = qkv + ((size_t)(0 * BATCH + b)) * L_TOK * DIM;
    const float* kbp = qkv + ((size_t)(1 * BATCH + b)) * L_TOK * DIM;
    const float* vbp = qkv + ((size_t)(2 * BATCH + b)) * L_TOK * DIM;
    const int chBase = head * HEAD_DIM;

    // ---------- stage K (row-major) and V^T (transposed) into LDS as bf16 ----------
    {
        const int t0 = tid * 2;                       // two consecutive tokens per thread
        const int l0 = ((t0 >> 3) * RES) + win * SPLIT + (t0 & 7);   // l1 = l0 + 1
        const float* k0 = kbp + (size_t)l0 * DIM + chBase;
        const float* k1 = k0 + DIM;
        const float* v0 = vbp + (size_t)l0 * DIM + chBase;
        const float* v1 = v0 + DIM;
        uint32_t* dk0 = sK + (uint32_t)t0 * KSTRIDE;
        uint32_t* dk1 = dk0 + KSTRIDE;
#pragma unroll
        for (int d = 0; d < HEAD_DIM; d += 2) {
            dk0[d >> 1] = pack_bf16_2(k0[d], k0[d + 1]);
            dk1[d >> 1] = pack_bf16_2(k1[d], k1[d + 1]);
        }
#pragma unroll
        for (int d = 0; d < HEAD_DIM; ++d)            // packed pair (t0, t0+1) per dword
            sVT[(uint32_t)d * VTSTRIDE + (uint32_t)tid] = pack_bf16_2(v0[d], v1[d]);
    }
    if (tid < HEAD_DIM) {
        const int ch = chBase + tid;
#pragma unroll
        for (int j = 0; j < 9; ++j) sW[tid * 10 + j] = conv_w[ch * 9 + j];
        sW[tid * 10 + 9] = conv_b[ch];
    }
    __syncthreads();

    // all-ones B fragment (bf16 1.0 splat): acc[4] = P @ 1 = softmax denominator
    Frag onef;
#pragma unroll
    for (int j = 0; j < 8; ++j) onef.u[j] = 0x3f803f80u;

    // ---------- per wave: 4 q-tiles of 16 rows, flash streaming over K ----------
    for (int qt = 0; qt < 4; ++qt) {
        const int qbase = wave * 64 + qt * 16;

        // Q A-fragments (2 chunks covering d=0..63), scaled by 1/sqrt(hd).
        // A layout: row M = laneLo; laneHi selects K-subgroups {0..15} vs {16..31}
        Frag qa[2];
        {
            const int t = qbase + laneLo;
            const int l = ((t >> 3) * RES) + win * SPLIT + (t & 7);
            const float* qrow = qb + (size_t)l * DIM + chBase;
#pragma unroll
            for (int c = 0; c < 2; ++c) {
                const int d0 = c * 32 + 16 * laneHi;
                const int d1 = c * 32 + 16 * laneHi + 8;
#pragma unroll
                for (int j = 0; j < 4; ++j) {
                    qa[c].u[j]     = pack_bf16_2(qrow[d0 + 2 * j] * QK_SCALE,
                                                 qrow[d0 + 2 * j + 1] * QK_SCALE);
                    qa[c].u[4 + j] = pack_bf16_2(qrow[d1 + 2 * j] * QK_SCALE,
                                                 qrow[d1 + 2 * j + 1] * QK_SCALE);
                }
            }
        }

        v8f acc[5];                         // 4 d-tiles of O  +  1 denominator tile
#pragma unroll
        for (int a = 0; a < 5; ++a)
#pragma unroll
            for (int r = 0; r < 8; ++r) acc[a][r] = 0.f;
        float rm[8];
#pragma unroll
        for (int r = 0; r < 8; ++r) rm[r] = -1e30f;

        for (int kb = 0; kb < 16; ++kb) {          // 32 key tokens per step
            // -------- scores: two 16x16 tiles, contraction over d in 2 chunks --------
            v8f s0, s1;
#pragma unroll
            for (int r = 0; r < 8; ++r) { s0[r] = 0.f; s1[r] = 0.f; }
#pragma unroll
            for (int c = 0; c < 2; ++c) {
                // B layout: lane = N + 16*(kd/16); per lane 16 consecutive d values
                const uint32_t off = (uint32_t)(c * 16 + 8 * laneHi);
                Frag bk0, bk1;
                {
                    const uint32_t* s = sK + (uint32_t)(kb * 32 + laneLo) * KSTRIDE + off;
                    bk0.x[0] = *(const u32x4*)s;
                    bk0.x[1] = *(const u32x4*)(s + 4);
                }
                {
                    const uint32_t* s = sK + (uint32_t)(kb * 32 + 16 + laneLo) * KSTRIDE + off;
                    bk1.x[0] = *(const u32x4*)s;
                    bk1.x[1] = *(const u32x4*)(s + 4);
                }
                s0 = __builtin_amdgcn_wmma_f32_16x16x32_bf16(false, qa[c].v, false, bk0.v,
                                                             (short)0, s0, false, false);
                s1 = __builtin_amdgcn_wmma_f32_16x16x32_bf16(false, qa[c].v, false, bk1.v,
                                                             (short)0, s1, false, false);
            }

            // -------- online softmax update (rows m = r + 8*laneHi) --------
            float resc[8];
#pragma unroll
            for (int r = 0; r < 8; ++r) {
                const float mx   = rowmax16(vmaxf(s0[r], s1[r]));
                const float mnew = vmaxf(rm[r], mx);
                resc[r] = __expf(rm[r] - mnew);
                rm[r]   = mnew;
                s0[r]   = __expf(s0[r] - mnew);
                s1[r]   = __expf(s1[r] - mnew);
            }
#pragma unroll
            for (int a = 0; a < 5; ++a)             // rescale O and denominator together
#pragma unroll
                for (int r = 0; r < 8; ++r) acc[a][r] *= resc[r];

            // -------- C-layout P -> A-fragment layout via per-wave LDS bounce --------
            {
                uint16_t* ps = (uint16_t*)(sP + wave * 256);   // [16 rows][32 cols] bf16
#pragma unroll
                for (int r = 0; r < 8; ++r) {
                    const int m = r + 8 * laneHi;
                    const uint32_t pr = pack_bf16_2(s0[r], s1[r]);
                    ps[m * 32 + laneLo]      = (uint16_t)pr;          // ds_store_b16
                    ps[m * 32 + 16 + laneLo] = (uint16_t)(pr >> 16);  // ds_store_b16_d16_hi
                }
            }
            asm volatile("" ::: "memory");       // keep LDS stores before the reads
            __builtin_amdgcn_wave_barrier();     // same-wave LDS is in-order on CDNA5

            Frag pf;
            {
                const uint32_t* s = sP + wave * 256 + laneLo * 16 + 4 * laneHi;
                pf.x[0] = *(const u32x4*)s;        // kk =      8*laneHi .. +7
                pf.x[1] = *(const u32x4*)(s + 8);  // kk = 16 + 8*laneHi .. +7
            }
            asm volatile("" ::: "memory");
            __builtin_amdgcn_wave_barrier();

            // -------- O += P @ V (4 d-tiles) and denom += P @ 1 --------
#pragma unroll
            for (int dt = 0; dt < 4; ++dt) {
                Frag vf;   // B layout from V^T: lane holds d=dt*16+laneLo, tokens {8h..8h+7, 16+8h..16+8h+7}
                const uint32_t* s = sVT + (uint32_t)(dt * 16 + laneLo) * VTSTRIDE
                                        + (uint32_t)(kb * 16 + 4 * laneHi);
                vf.x[0] = *(const u32x4*)s;
                vf.x[1] = *(const u32x4*)(s + 8);
                acc[dt] = __builtin_amdgcn_wmma_f32_16x16x32_bf16(false, pf.v, false, vf.v,
                                                                  (short)0, acc[dt], false, false);
            }
            acc[4] = __builtin_amdgcn_wmma_f32_16x16x32_bf16(false, pf.v, false, onef.v,
                                                             (short)0, acc[4], false, false);
        }

        // ---------- epilogue: O * rcp(denom) + LePE depthwise 3x3 conv ----------
        float inv[8];
#pragma unroll
        for (int r = 0; r < 8; ++r) inv[r] = __builtin_amdgcn_rcpf(acc[4][r]);

#pragma unroll
        for (int dt = 0; dt < 4; ++dt) {
            const int d  = dt * 16 + laneLo;      // fixed channel per lane per d-tile
            const int ch = chBase + d;
            const float* wv = sW + d * 10;
            float w[9], bias;
#pragma unroll
            for (int j = 0; j < 9; ++j) w[j] = wv[j];
            bias = wv[9];
#pragma unroll
            for (int r = 0; r < 8; ++r) {
                const int m  = r + 8 * laneHi;
                const int t  = qbase + m;
                const int hs = t >> 3, wsp = t & 7;
                float lepe = bias;
#pragma unroll
                for (int dy = -1; dy <= 1; ++dy) {
                    const int y = hs + dy;
                    if ((unsigned)y < (unsigned)RES) {
#pragma unroll
                        for (int dx = -1; dx <= 1; ++dx) {
                            const int x = wsp + dx;
                            if ((unsigned)x < (unsigned)SPLIT) {
                                lepe += w[(dy + 1) * 3 + (dx + 1)] *
                                        vbp[(size_t)(y * RES + win * SPLIT + x) * DIM + ch];
                            }
                        }
                    }
                }
                const float o = acc[dt][r] * inv[r] + lepe;
                const size_t l = (size_t)(hs * RES + win * SPLIT + wsp);
                out[((size_t)b * L_TOK + l) * DIM + ch] = o;
            }
        }
    }
}

extern "C" void kernel_launch(void* const* d_in, const int* in_sizes, int n_in,
                              void* d_out, int out_size, void* d_ws, size_t ws_size,
                              hipStream_t stream) {
    (void)in_sizes; (void)n_in; (void)out_size; (void)d_ws; (void)ws_size;
    const float* qkv    = (const float*)d_in[0];
    const float* conv_w = (const float*)d_in[1];
    const float* conv_b = (const float*)d_in[2];
    float* out = (float*)d_out;

    hipFuncSetAttribute((const void*)cswin_lepe_attn,
                        hipFuncAttributeMaxDynamicSharedMemorySize,
                        (int)SMEM_BYTES);
    dim3 grid(BATCH * NWIN * NUM_HEADS);   // 512 (window, head) tiles
    cswin_lepe_attn<<<grid, 256, SMEM_BYTES, stream>>>(qkv, conv_w, conv_b, out);
}
